// RWKVBlock_28750511079444
// MI455X (gfx1250) — matrix operand order repaired and checked by reading, not compiled, not run on hardware
//
#include <hip/hip_runtime.h>

// ---------------------------------------------------------------------------
// RWKV-v6 block forward for MI455X (gfx1250, wave32, WMMA).
// B=2, T=512, E=D=1024, H=16, Sh=64, F=4096, CHUNK=64.
// Big GEMMs run through v_wmma_f32_16x16x32_bf16 with f32 accumulation.
// ---------------------------------------------------------------------------

typedef __attribute__((ext_vector_type(16))) __bf16 v16bf;
typedef __attribute__((ext_vector_type(8)))  float  v8f;

union WFrag { unsigned int u[8]; uint4 q[2]; v16bf v; };

static constexpr int cB  = 2;
static constexpr int cT  = 512;
static constexpr int cE  = 1024;
static constexpr int cD  = 1024;
static constexpr int cH  = 16;
static constexpr int cSh = 64;
static constexpr int cF  = 4096;
static constexpr int cBT = cB * cT;          // 1024 tokens
static constexpr int cNCH = cT / 64;         // 8 chunks

// ---------------------------------------------------------------------------
// Generic bf16 GEMM: C(MxN,f32) = A(MxK,bf16,row-major) @ B(KxN,bf16,row-major)
// Block tile 128x128, BK=32, 256 threads = 8 waves arranged 2(M) x 4(N).
// Wave tile 64x32 = 4x2 WMMA tiles -> 8 v_wmma per K-slab per wave.
// B is staged into LDS pre-packed as (k-pair, n) dwords so every fragment
// load in the hot loop is a plain dword/b128 LDS read.
// ---------------------------------------------------------------------------
__global__ __launch_bounds__(256)
void gemm_bf16_kernel(const __bf16* __restrict__ A, const __bf16* __restrict__ B,
                      float* __restrict__ C, int M, int N, int K) {
  __shared__ __align__(16) __bf16 As[128 * 32];       // 8 KB
  __shared__ __align__(16) unsigned int Bp[16 * 128]; // 8 KB, packed k-pairs

  const int tid  = threadIdx.x;
  const int wave = tid >> 5;
  const int lane = tid & 31;
  const int lr   = lane & 15;     // row/col within 16
  const int half = lane >> 4;     // 0/1
  const int wm   = wave >> 2;     // wave row  (2 x 64 rows = 128)
  const int wn   = wave & 3;      // wave col  (4 x 32 cols = 128)
  const int m0   = blockIdx.y * 128;
  const int n0   = blockIdx.x * 128;

  v8f acc[4][2];
  for (int i = 0; i < 4; i++)
    for (int j = 0; j < 2; j++)
      for (int e = 0; e < 8; e++) acc[i][j][e] = 0.0f;

  for (int k0 = 0; k0 < K; k0 += 32) {
    __syncthreads();
    // ---- stage A tile 128x32 (512 uint4, 2 per thread) ----
    for (int q = tid; q < 512; q += 256) {
      const int row = q >> 2, seg = (q & 3) * 8;  // 8 bf16 per uint4
      *(uint4*)(As + row * 32 + seg) =
          *(const uint4*)(A + (size_t)(m0 + row) * K + k0 + seg);
    }
    // ---- stage B tile 32x128 packed into k-pair dwords ----
    for (int q = tid; q < 512; q += 256) {
      const int kp = q >> 5, n4 = (q & 31) * 4;   // 4 n-positions per uint4
      const int k = k0 + 2 * kp;
      const uint2 a = *(const uint2*)(B + (size_t)k * N + n0 + n4);
      const uint2 b = *(const uint2*)(B + (size_t)(k + 1) * N + n0 + n4);
      uint4 d;
      d.x = (a.x & 0xffffu) | (b.x << 16);
      d.y = (a.x >> 16)     | (b.x & 0xffff0000u);
      d.z = (a.y & 0xffffu) | (b.y << 16);
      d.w = (a.y >> 16)     | (b.y & 0xffff0000u);
      *(uint4*)(Bp + kp * 128 + n4) = d;
    }
    if (k0 + 32 < K) {
      __builtin_prefetch(A + (size_t)(m0 + (tid >> 2)) * K + k0 + 32, 0, 0);
      __builtin_prefetch(B + (size_t)(k0 + 32 + (tid >> 3)) * N + n0, 0, 0);
    }
    __syncthreads();

    // ---- fragments ----
    // A (ISA 16-bit A layout): lane lr = row m, element g covers
    // k = (g<4?0:16) + half*8 + (g&3)*2 (+1). Two b128 LDS loads per frag.
    WFrag af[4];
    for (int ta = 0; ta < 4; ta++) {
      const int am = wm * 64 + ta * 16 + lr;
      af[ta].q[0] = *(const uint4*)(As + am * 32 + half * 8);
      af[ta].q[1] = *(const uint4*)(As + am * 32 + 16 + half * 8);
    }
    // B: lane lr = column n; element g is k-pair kp = half*4 + g + (g&4).
    WFrag bf[2];
    for (int tb = 0; tb < 2; tb++) {
      const int bn = wn * 32 + tb * 16 + lr;
      for (int g = 0; g < 8; g++)
        bf[tb].u[g] = Bp[(half * 4 + g + (g & 4)) * 128 + bn];
    }
    for (int ta = 0; ta < 4; ta++)
      for (int tb = 0; tb < 2; tb++)
        acc[ta][tb] = __builtin_amdgcn_wmma_f32_16x16x32_bf16(
            false, af[ta].v, false, bf[tb].v, (short)0, acc[ta][tb], false, false);
  }

  // C/D layout: lane lr = n, VGPR r -> m = r + 8*half
  for (int ta = 0; ta < 4; ta++)
    for (int tb = 0; tb < 2; tb++) {
      const int gm = m0 + wm * 64 + ta * 16;
      const int gn = n0 + wn * 32 + tb * 16 + lr;
      for (int r = 0; r < 8; r++) {
        const int m = gm + r + half * 8;
        C[(size_t)m * N + gn] = acc[ta][tb][r];
      }
    }
}

// ---------------------------------------------------------------------------
// Block-wide LayerNorm stats helper (256 threads, 1024 elements)
// ---------------------------------------------------------------------------
__device__ inline void ln_stats(const float* __restrict__ xg, float* red, int tid,
                                float& mean, float& rstd) {
  float s = 0.f, sq = 0.f;
  for (int e = tid; e < cE; e += 256) { float v = xg[e]; s += v; sq += v * v; }
  red[tid] = s; red[256 + tid] = sq;
  __syncthreads();
  for (int st = 128; st > 0; st >>= 1) {
    if (tid < st) { red[tid] += red[tid + st]; red[256 + tid] += red[256 + tid + st]; }
    __syncthreads();
  }
  mean = red[0] * (1.0f / cE);
  float var = red[256] * (1.0f / cE) - mean * mean;
  rstd = rsqrtf(var + 1e-5f);
  __syncthreads();
}

// ---------------------------------------------------------------------------
// mix1: ln1 + time-shift + 5-way token mix (faithful cross-batch reshape
// scramble) + decay path. One block per timestep t, handles both batches.
// ---------------------------------------------------------------------------
__global__ __launch_bounds__(256)
void mix1_kernel(const float* __restrict__ x,
                 const float* __restrict__ maa_x,
                 const float* __restrict__ maa_w, const float* __restrict__ maa_k,
                 const float* __restrict__ maa_v, const float* __restrict__ maa_r,
                 const float* __restrict__ maa_g,
                 const float* __restrict__ tdecay,
                 const float* __restrict__ W1, const float* __restrict__ W2,
                 const float* __restrict__ dw1, const float* __restrict__ dw2,
                 const float* __restrict__ ln1_s, const float* __restrict__ ln1_b,
                 __bf16* __restrict__ xk_b, __bf16* __restrict__ xv_b,
                 __bf16* __restrict__ xr_b, __bf16* __restrict__ xg_b,
                 float* __restrict__ wdec) {
  __shared__ float red[512];
  __shared__ float xn[2][cE], sx[2][cE], xm[2][cE];
  __shared__ float t5s[2][160];
  __shared__ float td[2][64];

  const int t = blockIdx.x, tid = threadIdx.x;

  for (int b = 0; b < 2; b++) {
    const float* xc = x + ((size_t)(b * cT + t)) * cE;
    float mean, rstd, meanp = 0.f, rstdp = 0.f;
    ln_stats(xc, red, tid, mean, rstd);
    if (t > 0) ln_stats(xc - cE, red, tid, meanp, rstdp);
    for (int e = tid; e < cE; e += 256) {
      float s = ln1_s[e], bb = ln1_b[e];
      float cur = (xc[e] - mean) * rstd * s + bb;
      float prv = (t > 0) ? ((xc[e - cE] - meanp) * rstdp * s + bb) : 0.f;
      xn[b][e] = cur;
      float d = prv - cur;
      sx[b][e] = d;
      xm[b][e] = cur + d * maa_x[e];
    }
    __syncthreads();
    if (tid < 160) {
      float a = 0.f;
      for (int e = 0; e < cE; e++) a += xm[b][e] * W1[(size_t)e * 160 + tid];
      t5s[b][tid] = tanhf(a);
    }
    __syncthreads();
  }

  // m = einsum then reshape (B,5,T,E)->(5,B,T,E): output group go for batch b
  // reads original (bo, gi) with idx = go*2+b, bo = idx/5, gi = idx%5.
  for (int b = 0; b < 2; b++) {
    const size_t obase = ((size_t)(b * cT + t)) * cE;
    for (int e = tid; e < cE; e += 256) {
      float m[5];
      for (int go = 0; go < 5; go++) {
        const int idx = go * 2 + b;
        const int bo = idx / 5, gi = idx % 5;
        const float* w2g = W2 + (size_t)gi * 32 * cE + e;
        const float* t5p = t5s[bo] + gi * 32;
        float a = 0.f;
        for (int c = 0; c < 32; c++) a += t5p[c] * w2g[(size_t)c * cE];
        m[go] = a;
      }
      float xnv = xn[b][e], sxv = sx[b][e];
      xk_b[obase + e] = (__bf16)(xnv + sxv * (maa_k[e] + m[1]));
      xv_b[obase + e] = (__bf16)(xnv + sxv * (maa_v[e] + m[2]));
      xr_b[obase + e] = (__bf16)(xnv + sxv * (maa_r[e] + m[3]));
      xg_b[obase + e] = (__bf16)(xnv + sxv * (maa_g[e] + m[4]));
      xm[b][e] = xnv + sxv * (maa_w[e] + m[0]);  // xw, reused for decay
    }
  }
  __syncthreads();
  if (tid < 128) {
    const int b = tid >> 6, j = tid & 63;
    float a = 0.f;
    for (int e = 0; e < cE; e++) a += xm[b][e] * dw1[(size_t)e * 64 + j];
    td[b][j] = tanhf(a);
  }
  __syncthreads();
  for (int b = 0; b < 2; b++) {
    const size_t obase = ((size_t)(b * cT + t)) * cE;
    for (int e = tid; e < cE; e += 256) {
      float off = 0.f;
      for (int j = 0; j < 64; j++) off += td[b][j] * dw2[(size_t)j * cE + e];
      float z = -(tdecay[e] + off);
      wdec[obase + e] = (z > 30.f) ? z : log1pf(__expf(z));   // softplus
    }
  }
}

// ---------------------------------------------------------------------------
// WKV chunkwise recurrence. One block per (b,h) lane; S(64x64) lives in LDS.
// ---------------------------------------------------------------------------
__global__ __launch_bounds__(256)
void wkv_kernel(const float* __restrict__ rF, const float* __restrict__ kF,
                const float* __restrict__ vF, const float* __restrict__ wdec,
                const float* __restrict__ faaaa, const float* __restrict__ state_in,
                float* __restrict__ yF, float* __restrict__ state_out) {
  extern __shared__ float sm[];
  float* S   = sm;                  // 64*65
  float* Ac  = S + 64 * 65;         // 64*65  (A_intra cumsum)
  float* rd  = Ac + 64 * 65;        // 64*65
  float* kd  = rd + 64 * 65;        // 64*65
  float* P   = kd + 64 * 65;        // 64*65
  float* ych = P + 64 * 65;         // 64*64  (inter partial)
  float* av  = ych + 64 * 64;
  float* bv  = av + 64;
  float* qv  = bv + 64;
  float* vv  = qv + 64;
  float* ud  = vv + 64;

  const int bh = blockIdx.x, b = bh >> 4, h = bh & 15;
  const int tid = threadIdx.x;

  for (int idx = tid; idx < 4096; idx += 256) {
    const int j = idx >> 6, kk = idx & 63;
    S[j * 65 + kk] = state_in[(size_t)bh * 4096 + idx];
  }
  __syncthreads();

  for (int n = 0; n < cNCH; n++) {
    if (tid < 64) {
      const int j = tid;
      float run = 0.f;
      for (int c = 0; c < 64; c++) {
        const int t = n * 64 + c;
        float wv = wdec[((size_t)(b * cT + t)) * cD + h * 64 + j];
        wv = fminf(fmaxf(wv, 1e-6f), 1.0f);
        run += __logf(wv);
        Ac[c * 65 + j] = run;
      }
    }
    __syncthreads();

    for (int c = 0; c < 64; c++) {
      const int t = n * 64 + c;
      const size_t gb = ((size_t)(b * cT + t)) * cD + h * 64;
      if (tid < 64) {
        const int j = tid;
        const float acj  = Ac[c * 65 + j];
        const float prev = c ? Ac[(c - 1) * 65 + j] : 0.f;
        const float wl   = acj - prev;
        const float atot = Ac[63 * 65 + j];
        av[j] = __expf(acj);
        bv[j] = __expf(atot - acj) * kF[gb + j];
        qv[j] = rF[gb + j] * __expf(acj - wl);
        vv[j] = vF[gb + j];
      }
      __syncthreads();
      if (tid < 64) {
        const int kk = tid;
        float a = 0.f;
        for (int j = 0; j < 64; j++) a += qv[j] * S[j * 65 + kk];
        ych[c * 64 + kk] = a;
      }
      __syncthreads();
      {
        const int j = tid >> 2, kb = (tid & 3) * 16;
        const float aj = av[j], bj = bv[j];
        for (int q = 0; q < 16; q++) {
          const int idx = j * 65 + kb + q;
          S[idx] = aj * S[idx] + bj * vv[kb + q];
        }
      }
      __syncthreads();
    }

    for (int idx = tid; idx < 4096; idx += 256) {
      const int c = idx >> 6, j = idx & 63;
      const int t = n * 64 + c;
      const size_t gb = ((size_t)(b * cT + t)) * cD + h * 64;
      const float amid  = Ac[32 * 65 + j];
      const float acj   = Ac[c * 65 + j];
      const float aprev = c ? Ac[(c - 1) * 65 + j] : 0.f;
      rd[c * 65 + j] = rF[gb + j] * __expf(aprev - amid);
      kd[c * 65 + j] = kF[gb + j] * __expf(amid - acj);
    }
    if (tid < 64) {
      const int c = tid;
      const size_t gb = ((size_t)(b * cT + n * 64 + c)) * cD + h * 64;
      float a = 0.f;
      for (int j = 0; j < 64; j++) a += rF[gb + j] * faaaa[h * 64 + j] * kF[gb + j];
      ud[c] = a;
    }
    __syncthreads();
    for (int idx = tid; idx < 4096; idx += 256) {
      const int c = idx >> 6, c2 = idx & 63;
      float a = 0.f;
      for (int j = 0; j < 64; j++) a += rd[c * 65 + j] * kd[c2 * 65 + j];
      P[c * 65 + c2] = a;
    }
    __syncthreads();
    for (int idx = tid; idx < 4096; idx += 256) {
      const int c = idx >> 6, kk = idx & 63;
      const int t = n * 64 + c;
      float a = ud[c] * vF[((size_t)(b * cT + t)) * cD + h * 64 + kk];
      for (int c2 = 0; c2 < c; c2++)
        a += P[c * 65 + c2] * vF[((size_t)(b * cT + n * 64 + c2)) * cD + h * 64 + kk];
      yF[((size_t)(b * cT + t)) * cD + h * 64 + kk] = ych[c * 64 + kk] + a;
    }
    __syncthreads();
  }

  for (int idx = tid; idx < 4096; idx += 256) {
    const int j = idx >> 6, kk = idx & 63;
    state_out[(size_t)bh * 4096 + idx] = S[j * 65 + kk];
  }
}

// ---------------------------------------------------------------------------
// GroupNorm: per-head stats over (B*T rows, 64 channels), then apply + gate.
// ---------------------------------------------------------------------------
__global__ __launch_bounds__(256)
void gn_stats_kernel(const float* __restrict__ yF, float* __restrict__ stats) {
  __shared__ float red[512];
  const int h = blockIdx.x, tid = threadIdx.x;
  float s = 0.f, sq = 0.f;
  for (int idx = tid; idx < cBT * 64; idx += 256) {
    const int row = idx >> 6, cc = idx & 63;
    float v = yF[(size_t)row * cD + h * 64 + cc];
    s += v; sq += v * v;
  }
  red[tid] = s; red[256 + tid] = sq;
  __syncthreads();
  for (int st = 128; st > 0; st >>= 1) {
    if (tid < st) { red[tid] += red[tid + st]; red[256 + tid] += red[256 + tid + st]; }
    __syncthreads();
  }
  if (tid == 0) {
    const float nval = (float)(cBT * 64);
    float mean = red[0] / nval;
    float var  = red[256] / nval - mean * mean;
    stats[2 * h]     = mean;
    stats[2 * h + 1] = rsqrtf(var + 1e-5f);
  }
}

__global__ void gn_apply_kernel(const float* __restrict__ yF, const float* __restrict__ gF,
                                const float* __restrict__ stats,
                                const float* __restrict__ scale, const float* __restrict__ bias,
                                __bf16* __restrict__ ygb, int n) {
  const int i = blockIdx.x * 256 + threadIdx.x;
  if (i >= n) return;
  const int d = i & (cD - 1);
  const int h = d >> 6;
  float yn = (yF[i] - stats[2 * h]) * stats[2 * h + 1] * scale[d] + bias[d];
  ygb[i] = (__bf16)(yn * gF[i]);
}

// ---------------------------------------------------------------------------
// mix2: ln2 + time-shift channel-mix inputs (bf16 out).
// ---------------------------------------------------------------------------
__global__ __launch_bounds__(256)
void mix2_kernel(const float* __restrict__ x1,
                 const float* __restrict__ maa_k, const float* __restrict__ maa_r,
                 const float* __restrict__ ln2_s, const float* __restrict__ ln2_b,
                 __bf16* __restrict__ xkc, __bf16* __restrict__ xrc) {
  __shared__ float red[512];
  const int blk = blockIdx.x, tid = threadIdx.x;
  const int t = blk & (cT - 1);
  const float* xc = x1 + (size_t)blk * cE;
  float mean, rstd, meanp = 0.f, rstdp = 0.f;
  ln_stats(xc, red, tid, mean, rstd);
  if (t > 0) ln_stats(xc - cE, red, tid, meanp, rstdp);
  for (int e = tid; e < cE; e += 256) {
    float s = ln2_s[e], bb = ln2_b[e];
    float cur = (xc[e] - mean) * rstd * s + bb;
    float prv = (t > 0) ? ((xc[e - cE] - meanp) * rstdp * s + bb) : 0.f;
    float d = prv - cur;
    xkc[(size_t)blk * cE + e] = (__bf16)(cur + d * maa_k[e]);
    xrc[(size_t)blk * cE + e] = (__bf16)(cur + d * maa_r[e]);
  }
}

// ---------------------------------------------------------------------------
// Small elementwise kernels
// ---------------------------------------------------------------------------
__global__ void f32_to_bf16_kernel(const float* __restrict__ in, __bf16* __restrict__ out, int n) {
  const int i = blockIdx.x * 256 + threadIdx.x;
  if (i < n) out[i] = (__bf16)in[i];
}

__global__ void silu_kernel(float* __restrict__ g, int n) {
  const int i = blockIdx.x * 256 + threadIdx.x;
  if (i < n) { float v = g[i]; g[i] = v / (1.0f + __expf(-v)); }
}

__global__ void add2_kernel(const float* __restrict__ a, const float* __restrict__ b,
                            float* __restrict__ out, int n) {
  const int i = blockIdx.x * 256 + threadIdx.x;
  if (i < n) out[i] = a[i] + b[i];
}

__global__ void relusq_kernel(const float* __restrict__ in, __bf16* __restrict__ out, int n) {
  const int i = blockIdx.x * 256 + threadIdx.x;
  if (i < n) { float v = fmaxf(in[i], 0.f); out[i] = (__bf16)(v * v); }
}

__global__ void final_kernel(const float* __restrict__ x1, const float* __restrict__ rr,
                             const float* __restrict__ kv, float* __restrict__ out, int n) {
  const int i = blockIdx.x * 256 + threadIdx.x;
  if (i < n) out[i] = x1[i] + (1.0f / (1.0f + __expf(-rr[i]))) * kv[i];
}

// ---------------------------------------------------------------------------
// Host-side orchestration
// ---------------------------------------------------------------------------
extern "C" void kernel_launch(void* const* d_in, const int* in_sizes, int n_in,
                              void* d_out, int out_size, void* d_ws, size_t ws_size,
                              hipStream_t stream) {
  (void)in_sizes; (void)n_in; (void)out_size; (void)ws_size;

  const float* x        = (const float*)d_in[0];
  const float* state_in = (const float*)d_in[1];
  const float* maa_x    = (const float*)d_in[2];
  const float* maa_w    = (const float*)d_in[3];
  const float* maa_k    = (const float*)d_in[4];
  const float* maa_v    = (const float*)d_in[5];
  const float* maa_r    = (const float*)d_in[6];
  const float* maa_g    = (const float*)d_in[7];
  const float* tdecay   = (const float*)d_in[8];
  const float* faaaa    = (const float*)d_in[9];
  const float* W1       = (const float*)d_in[10];
  const float* W2       = (const float*)d_in[11];
  const float* dw1      = (const float*)d_in[12];
  const float* dw2      = (const float*)d_in[13];
  const float* Wr       = (const float*)d_in[14];
  const float* Wk       = (const float*)d_in[15];
  const float* Wv       = (const float*)d_in[16];
  const float* Wg       = (const float*)d_in[17];
  const float* Wo       = (const float*)d_in[18];
  const float* lnx_s    = (const float*)d_in[19];
  const float* lnx_b    = (const float*)d_in[20];
  const float* maa_k_ch = (const float*)d_in[21];
  const float* maa_r_ch = (const float*)d_in[22];
  const float* Wk_ch    = (const float*)d_in[23];
  const float* Wv_ch    = (const float*)d_in[24];
  const float* Wr_ch    = (const float*)d_in[25];
  const float* ln1_s    = (const float*)d_in[26];
  const float* ln1_b    = (const float*)d_in[27];
  const float* ln2_s    = (const float*)d_in[28];
  const float* ln2_b    = (const float*)d_in[29];

  float* out       = (float*)d_out;                    // (B,T,E)
  float* state_out = out + (size_t)cBT * cE;           // (B,H,Sh,Sh)

  char* ws = (char*)d_ws;
  size_t off = 0;
  auto alloc = [&](size_t bytes) -> char* {
    char* p = ws + off;
    off += (bytes + 255) & ~(size_t)255;
    return p;
  };
  const size_t ED = (size_t)cE * cD, EF = (size_t)cE * cF, BTE = (size_t)cBT * cE,
               BTF = (size_t)cBT * cF;

  __bf16* Wr_b   = (__bf16*)alloc(ED * 2);
  __bf16* Wk_b   = (__bf16*)alloc(ED * 2);
  __bf16* Wv_b   = (__bf16*)alloc(ED * 2);
  __bf16* Wg_b   = (__bf16*)alloc(ED * 2);
  __bf16* Wo_b   = (__bf16*)alloc(ED * 2);
  __bf16* Wkch_b = (__bf16*)alloc(EF * 2);
  __bf16* Wvch_b = (__bf16*)alloc(EF * 2);
  __bf16* Wrch_b = (__bf16*)alloc(ED * 2);
  __bf16* xk_b   = (__bf16*)alloc(BTE * 2);
  __bf16* xv_b   = (__bf16*)alloc(BTE * 2);
  __bf16* xr_b   = (__bf16*)alloc(BTE * 2);
  __bf16* xg_b   = (__bf16*)alloc(BTE * 2);
  __bf16* ygb    = (__bf16*)alloc(BTE * 2);
  __bf16* xkc_b  = (__bf16*)alloc(BTE * 2);
  __bf16* xrc_b  = (__bf16*)alloc(BTE * 2);
  __bf16* kk_b   = (__bf16*)alloc(BTF * 2);
  float* wdec  = (float*)alloc(BTE * 4);
  float* rF    = (float*)alloc(BTE * 4);
  float* kF    = (float*)alloc(BTE * 4);
  float* vF    = (float*)alloc(BTE * 4);
  float* gF    = (float*)alloc(BTE * 4);
  float* yF    = (float*)alloc(BTE * 4);
  float* attF  = (float*)alloc(BTE * 4);
  float* x1F   = (float*)alloc(BTE * 4);
  float* kkraw = (float*)alloc(BTF * 4);
  float* kvF   = (float*)alloc(BTE * 4);
  float* rrF   = (float*)alloc(BTE * 4);
  float* stats = (float*)alloc(2 * cH * 4);

  auto ewg = [](size_t n) { return dim3((unsigned)((n + 255) / 256)); };
  auto cvt = [&](const float* src, __bf16* dst, size_t n) {
    f32_to_bf16_kernel<<<ewg(n), 256, 0, stream>>>(src, dst, (int)n);
  };
  auto gemm = [&](const __bf16* A, const __bf16* Bm, float* C, int M, int N, int K) {
    dim3 grid(N / 128, M / 128);
    gemm_bf16_kernel<<<grid, 256, 0, stream>>>(A, Bm, C, M, N, K);
  };

  // 1) weights -> bf16
  cvt(Wr, Wr_b, ED);   cvt(Wk, Wk_b, ED);   cvt(Wv, Wv_b, ED);   cvt(Wg, Wg_b, ED);
  cvt(Wo, Wo_b, ED);   cvt(Wk_ch, Wkch_b, EF); cvt(Wv_ch, Wvch_b, EF); cvt(Wr_ch, Wrch_b, ED);

  // 2) ln1 + token mix + decay
  mix1_kernel<<<cT, 256, 0, stream>>>(x, maa_x, maa_w, maa_k, maa_v, maa_r, maa_g,
                                      tdecay, W1, W2, dw1, dw2, ln1_s, ln1_b,
                                      xk_b, xv_b, xr_b, xg_b, wdec);

  // 3) projections (WMMA GEMMs)
  gemm(xr_b, Wr_b, rF, cBT, cD, cE);
  gemm(xk_b, Wk_b, kF, cBT, cD, cE);
  gemm(xv_b, Wv_b, vF, cBT, cD, cE);
  gemm(xg_b, Wg_b, gF, cBT, cD, cE);
  silu_kernel<<<ewg(BTE), 256, 0, stream>>>(gF, (int)BTE);

  // 4) WKV chunkwise recurrence (writes new_state into d_out tail)
  const size_t wkv_smem = (size_t)(64 * 65 * 5 + 64 * 64 + 5 * 64) * sizeof(float);
  wkv_kernel<<<cB * cH, 256, wkv_smem, stream>>>(rF, kF, vF, wdec, faaaa, state_in,
                                                 yF, state_out);

  // 5) group norm + gate, output projection + residual
  gn_stats_kernel<<<cH, 256, 0, stream>>>(yF, stats);
  gn_apply_kernel<<<ewg(BTE), 256, 0, stream>>>(yF, gF, stats, lnx_s, lnx_b, ygb, (int)BTE);
  gemm(ygb, Wo_b, attF, cBT, cE, cD);
  add2_kernel<<<ewg(BTE), 256, 0, stream>>>(x, attF, x1F, (int)BTE);

  // 6) channel mixing
  mix2_kernel<<<cBT, 256, 0, stream>>>(x1F, maa_k_ch, maa_r_ch, ln2_s, ln2_b, xkc_b, xrc_b);
  gemm(xkc_b, Wkch_b, kkraw, cBT, cF, cE);
  relusq_kernel<<<ewg(BTF), 256, 0, stream>>>(kkraw, kk_b, (int)BTF);
  gemm(kk_b, Wvch_b, kvF, cBT, cE, cF);
  gemm(xrc_b, Wrch_b, rrF, cBT, cE, cE);
  final_kernel<<<ewg(BTE), 256, 0, stream>>>(x1F, rrF, kvF, out, (int)BTE);
}
